// ContrastivePromptLearning_60988535603877
// MI455X (gfx1250) — compile-verified, matrix operand order checked by reading, not run on hardware
//
#include <hip/hip_runtime.h>
#include <hip/hip_bf16.h>
#include <math.h>

typedef __attribute__((ext_vector_type(16))) __bf16 v16bf;
typedef __attribute__((ext_vector_type(8)))  __bf16 v8bf;
typedef __attribute__((ext_vector_type(8)))  float  v8f;

#define EMBED_DIM 256
#define PROJ_DIM  128
#define NPTS      1024
#define NBATCH    32
#define M_TOTAL   (NBATCH * NPTS)   // 32768
#define BIGNEG    (-1000000000.0f)

// ---------------------------------------------------------------------------
// 16x32 bf16 operand fragment loader (A-matrix layout, also valid for B=W^T
// when W is row-major [N x K]):
//   lanes 0-15 : row = lane,    halfs 0-7 = K[k0..k0+7],   8-15 = K[k0+16..k0+23]
//   lanes 16-31: row = lane-16, halfs 0-7 = K[k0+8..k0+15],8-15 = K[k0+24..k0+31]
// Works for both global and LDS source pointers.
// ---------------------------------------------------------------------------
__device__ __forceinline__ v16bf load_frag16x32(const __bf16* __restrict__ base,
                                                int stride, int lane) {
  const int r  = lane & 15;
  const int hi = lane >> 4;
  const __bf16* p = base + (size_t)r * stride + hi * 8;
  v8bf lo  = *reinterpret_cast<const v8bf*>(p);
  v8bf hi8 = *reinterpret_cast<const v8bf*>(p + 16);
  v16bf f;
#pragma unroll
  for (int i = 0; i < 8; ++i) { f[i] = lo[i]; f[i + 8] = hi8[i]; }
  return f;
}

__device__ __forceinline__ float gelu_exact(float v) {
  return 0.5f * v * (1.0f + erff(v * 0.70710678118654752f));
}

// raw v_exp_f32: flush-to-zero underflow, no libm range fixup (exact for our use)
__device__ __forceinline__ float fast_exp2(float x) {
  return __builtin_amdgcn_exp2f(x);
}

// ---------------------------------------------------------------------------
// f32 -> bf16 convert (weights)
// ---------------------------------------------------------------------------
__global__ void f2bf_kernel(const float* __restrict__ s, __bf16* __restrict__ d, int n) {
  int i = blockIdx.x * blockDim.x + threadIdx.x;
  if (i < n) d[i] = (__bf16)s[i];
}

// ---------------------------------------------------------------------------
// Positional-encoding features -> bf16 [M_TOTAL x 256]
// ---------------------------------------------------------------------------
__global__ void pe_kernel(const float* __restrict__ pc, __bf16* __restrict__ out) {
  int idx = blockIdx.x * blockDim.x + threadIdx.x;   // M_TOTAL*256 threads
  int row = idx >> 8;
  int col = idx & 255;
  float cx = pc[row * 2 + 0] * (1.0f / 256.0f);
  float cy = pc[row * 2 + 1] * (1.0f / 256.0f);
  int q = col >> 6;
  int i = col & 63;
  // 1/dim_t = 10000^(-2i/64) = exp(-i * 2*ln(10000)/64)
  float invdim = __expf(-(float)i * 0.28782313662425572f);
  float ang = (q < 2 ? cx : cy) * 6.283185307179586f * invdim;
  float v = ((q & 1) == 0) ? __sinf(ang) : __cosf(ang);
  out[idx] = (__bf16)v;
}

// ---------------------------------------------------------------------------
// Tiled WMMA GEMM: C[M x NOUT] = A[M x 256] * W[NOUT x 256]^T + bias, + epilogue
// block = 256 threads = 8 waves (2 row-groups x 4 col-groups), tile 32 x NOUT.
// A tile staged via LDS; W fragments register-double-buffered so WMMAs overlap
// the next k-step's fetch.
// MODE 0: + type_emb[label]            -> bf16
// MODE 1: LayerNorm + exact GELU       -> bf16
// MODE 2: raw                          -> f32 (prompt_embed) + bf16
// MODE 3: row L2-normalize             -> bf16
// ---------------------------------------------------------------------------
template <int NOUT, int MODE>
__global__ void __launch_bounds__(256, 2)
gemm_ep_kernel(const __bf16* __restrict__ A,
               const __bf16* __restrict__ W,
               const float*  __restrict__ bias,
               const float*  __restrict__ ln_w,
               const float*  __restrict__ ln_b,
               const float*  __restrict__ type_emb,   // [2 x 256] (MODE 0)
               const int*    __restrict__ labels,     // [M]       (MODE 0)
               float*        __restrict__ out_f32,    // (MODE 2)
               __bf16*       __restrict__ out_bf) {
  constexpr int K  = 256;
  constexpr int NT = NOUT / 64;                // 16-wide tiles per wave
  const int tid  = threadIdx.x;
  const int wave = tid >> 5, lane = tid & 31;
  const int rowg = wave >> 2, colg = wave & 3;
  const int rowBlock = blockIdx.x * 32;

  // ---- stage A tile (32 x 256 bf16, 16KB) into LDS ----
  __shared__ __bf16 sA[32 * K];
  {
    const int r  = tid >> 3;            // 0..31
    const int c0 = (tid & 7) * 32;      // 0..224
    const v8bf* src = reinterpret_cast<const v8bf*>(A + (size_t)(rowBlock + r) * K + c0);
    v8bf* dst = reinterpret_cast<v8bf*>(&sA[r * K + c0]);
#pragma unroll
    for (int i = 0; i < 4; ++i) dst[i] = src[i];
  }
  __syncthreads();

  const __bf16* aBase = &sA[(rowg * 16) * K];

  v8f acc[NT];
#pragma unroll
  for (int t = 0; t < NT; ++t)
#pragma unroll
    for (int i = 0; i < 8; ++i) acc[t][i] = 0.0f;

  // ---- software-pipelined K loop: prefetch k+32 while computing k ----
  v16bf af = load_frag16x32(aBase, K, lane);
  v16bf bfr[NT];
#pragma unroll
  for (int t = 0; t < NT; ++t)
    bfr[t] = load_frag16x32(W + (size_t)(colg * (NOUT / 4) + t * 16) * K, K, lane);

#pragma unroll
  for (int k0 = 0; k0 < K - 32; k0 += 32) {
    v16bf afn = load_frag16x32(aBase + (k0 + 32), K, lane);
    v16bf bfn[NT];
#pragma unroll
    for (int t = 0; t < NT; ++t)
      bfn[t] = load_frag16x32(W + (size_t)(colg * (NOUT / 4) + t * 16) * K + (k0 + 32), K, lane);
#pragma unroll
    for (int t = 0; t < NT; ++t)
      acc[t] = __builtin_amdgcn_wmma_f32_16x16x32_bf16(
          false, af, false, bfr[t], (short)0, acc[t], false, false);
    af = afn;
#pragma unroll
    for (int t = 0; t < NT; ++t) bfr[t] = bfn[t];
  }
#pragma unroll
  for (int t = 0; t < NT; ++t)
    acc[t] = __builtin_amdgcn_wmma_f32_16x16x32_bf16(
        false, af, false, bfr[t], (short)0, acc[t], false, false);

  __syncthreads();

  // ---- accumulator -> LDS (full 32 x NOUT tile, row epilogues block-local) ----
  __shared__ float sOut[32 * NOUT];
  {
    const int rbase = rowg * 16 + (lane >> 4) * 8;
    const int nloc  = lane & 15;
#pragma unroll
    for (int t = 0; t < NT; ++t) {
      const int ncol = colg * (NOUT / 4) + t * 16 + nloc;
#pragma unroll
      for (int v = 0; v < 8; ++v)
        sOut[(rbase + v) * NOUT + ncol] = acc[t][v];
    }
  }
  __syncthreads();

  // ---- epilogue: 8 threads per row, NOUT/8 cols each ----
  constexpr int CPT = NOUT / 8;
  const int r = tid >> 3;
  const int j = tid & 7;
  const int grow = rowBlock + r;
  float* rowp = &sOut[r * NOUT];

  if (MODE == 0) {
    int lab = labels[grow];
    lab = lab < 0 ? 0 : (lab > 1 ? 1 : lab);
    const float* te = type_emb + lab * 256;
    for (int c = j * CPT; c < (j + 1) * CPT; ++c) {
      float v = rowp[c] + bias[c] + te[c];
      out_bf[(size_t)grow * NOUT + c] = (__bf16)v;
    }
  } else if (MODE == 1) {
    float s = 0.0f, ss = 0.0f;
    for (int c = j * CPT; c < (j + 1) * CPT; ++c) {
      float v = rowp[c] + bias[c];
      s += v; ss += v * v;
    }
#pragma unroll
    for (int off = 1; off < 8; off <<= 1) {
      s  += __shfl_xor(s, off, 32);
      ss += __shfl_xor(ss, off, 32);
    }
    const float mean = s * (1.0f / NOUT);
    const float var  = ss * (1.0f / NOUT) - mean * mean;
    const float rstd = rsqrtf(var + 1e-5f);
    for (int c = j * CPT; c < (j + 1) * CPT; ++c) {
      float v = (rowp[c] + bias[c] - mean) * rstd * ln_w[c] + ln_b[c];
      out_bf[(size_t)grow * NOUT + c] = (__bf16)gelu_exact(v);
    }
  } else if (MODE == 2) {
    for (int c = j * CPT; c < (j + 1) * CPT; ++c) {
      float v = rowp[c] + bias[c];
      out_f32[(size_t)grow * NOUT + c] = v;
      out_bf [(size_t)grow * NOUT + c] = (__bf16)v;
    }
  } else {  // MODE 3: L2 normalize
    float ss = 0.0f;
    for (int c = j * CPT; c < (j + 1) * CPT; ++c) {
      float v = rowp[c] + bias[c];
      ss += v * v;
    }
#pragma unroll
    for (int off = 1; off < 8; off <<= 1) ss += __shfl_xor(ss, off, 32);
    const float inv = 1.0f / fmaxf(sqrtf(ss), 1e-12f);
    for (int c = j * CPT; c < (j + 1) * CPT; ++c) {
      float v = (rowp[c] + bias[c]) * inv;
      out_bf[(size_t)grow * NOUT + c] = (__bf16)v;
    }
  }
}

// ---------------------------------------------------------------------------
// Per-batch similarity + streaming masked log-sum-exp loss.
// grid = (N/32, B); block = 256 thr = 8 waves (2 row-groups x 4 col-groups).
// A-fragments hoisted in VGPRs; B-fragments register-double-buffered across
// 64-wide column chunks.
//
// Numeric tricks (proj rows are unit-norm => s in [-1,1], logits <= invT):
//  * fixed-max LSE: term = raw_exp2(s*c1 + c0), c1 = invT*log2e, c0 = -c1
//    -> one v_fma + one raw v_exp per element; flush-to-zero underflow is
//    exact here (terms that small contribute 0 to the sum).
//  * self-mask folded out of the sum: diagonal similarity == 1 exactly, its
//    unmasked term is exp2(0) == 1, so subtract 1.0 from the reduced sum.
//    Self-exclusion still applied to the pos-max (tm) via one predicate.
// ---------------------------------------------------------------------------
__global__ void __launch_bounds__(256, 2)
sim_loss_kernel(const __bf16* __restrict__ proj,   // [B*N x 128] bf16 (L2-normed)
                const int*    __restrict__ labels, // [B*N]
                const float*  __restrict__ logT,
                float*        __restrict__ batch_acc) {  // [B]
  const int b = blockIdx.y;
  const int rowBlock = blockIdx.x * 32;
  const int tid  = threadIdx.x;
  const int wave = tid >> 5, lane = tid & 31;
  const int rowg = wave >> 2, colg = wave & 3;
  const int aRow = b * NPTS + rowBlock + rowg * 16;

  float temp = __expf(logT[0]);
  temp = fminf(fmaxf(temp, 0.01f), 1.0f);
  const float invT = 1.0f / temp;
  const float c1 = invT * 1.4426950408889634f;  // invT * log2(e)
  const float c0 = -c1;

  // hoist A fragments (16 rows x 128 K = 4 frags)
  v16bf af[4];
#pragma unroll
  for (int kk = 0; kk < 4; ++kk)
    af[kk] = load_frag16x32(proj + (size_t)aRow * PROJ_DIM + kk * 32, PROJ_DIM, lane);

  float sm[8], tm[8];
#pragma unroll
  for (int v = 0; v < 8; ++v) { sm[v] = 0.0f; tm[v] = 2.0f * BIGNEG; }

  const int half     = lane >> 4;
  const int nlocal   = lane & 15;
  const int growBase = rowBlock + rowg * 16 + half * 8;

  // prefetch chunk 0
  v16bf bfr[4];
#pragma unroll
  for (int kk = 0; kk < 4; ++kk)
    bfr[kk] = load_frag16x32(proj + (size_t)(b * NPTS + colg * 16) * PROJ_DIM + kk * 32,
                             PROJ_DIM, lane);

  for (int chunk = 0; chunk < NPTS; chunk += 64) {
    const int colTile = chunk + colg * 16;

    // prefetch next chunk's B fragments while this chunk's WMMAs run
    v16bf bfn[4];
    const bool more = (chunk + 64) < NPTS;
    if (more) {
#pragma unroll
      for (int kk = 0; kk < 4; ++kk)
        bfn[kk] = load_frag16x32(
            proj + (size_t)(b * NPTS + colTile + 64) * PROJ_DIM + kk * 32, PROJ_DIM, lane);
    }

    v8f acc;
#pragma unroll
    for (int i = 0; i < 8; ++i) acc[i] = 0.0f;
#pragma unroll
    for (int kk = 0; kk < 4; ++kk)
      acc = __builtin_amdgcn_wmma_f32_16x16x32_bf16(
          false, af[kk], false, bfr[kk], (short)0, acc, false, false);

    const int gcol   = colTile + nlocal;
    const int labc   = labels[b * NPTS + gcol];
    const bool isPos = (labc == 1);
#pragma unroll
    for (int v = 0; v < 8; ++v) {
      const float s = acc[v];
      sm[v] += fast_exp2(fmaf(s, c1, c0));                   // unmasked LSE term
      const bool excl = isPos & (gcol != (growBase + v));    // pos col, not self
      tm[v] = fmaxf(tm[v], excl ? s : 2.0f * BIGNEG);
    }

    if (more) {
#pragma unroll
      for (int kk = 0; kk < 4; ++kk) bfr[kk] = bfn[kk];
    }
  }

  // reduce over the 16 N-lanes (xor 1,2,4,8 stays within each 16-lane half)
#pragma unroll
  for (int off = 1; off < 16; off <<= 1) {
#pragma unroll
    for (int v = 0; v < 8; ++v) {
      sm[v] += __shfl_xor(sm[v], off, 32);
      tm[v]  = fmaxf(tm[v], __shfl_xor(tm[v], off, 32));
    }
  }

  __shared__ float sSm[32][4], sTm[32][4];
  if (nlocal == 0) {
#pragma unroll
    for (int v = 0; v < 8; ++v) {
      const int r = rowg * 16 + half * 8 + v;
      sSm[r][colg] = sm[v]; sTm[r][colg] = tm[v];
    }
  }
  __syncthreads();

  if (tid < 32) {
    float s = sSm[tid][0], t = sTm[tid][0];
#pragma unroll
    for (int c = 1; c < 4; ++c) {
      s += sSm[tid][c];
      t  = fmaxf(t, sTm[tid][c]);
    }
    const int grow = rowBlock + tid;
    if (labels[b * NPTS + grow] == 1) {
      // remove the (unmasked) diagonal term exp2(0)==1 from the sum
      const float lse = invT + __logf(s - 1.0f);
      const float lt  = t * invT;   // logits[argmax tscore] = max pos masked sim / temp
      atomicAdd(&batch_acc[b], lse - lt);
    }
  }
}

// ---------------------------------------------------------------------------
// Per-batch label counts (also zero-initializes batch accumulators)
// ---------------------------------------------------------------------------
__global__ void counts_kernel(const int* __restrict__ labels,
                              int* __restrict__ npos, int* __restrict__ nneg,
                              float* __restrict__ acc) {
  const int b = blockIdx.x;
  __shared__ int sp[256], sn[256];
  int p = 0, n = 0;
  for (int i = threadIdx.x; i < NPTS; i += 256) {
    const int l = labels[b * NPTS + i];
    p += (l == 1);
    n += (l == 0);
  }
  sp[threadIdx.x] = p; sn[threadIdx.x] = n;
  __syncthreads();
  for (int off = 128; off > 0; off >>= 1) {
    if (threadIdx.x < off) {
      sp[threadIdx.x] += sp[threadIdx.x + off];
      sn[threadIdx.x] += sn[threadIdx.x + off];
    }
    __syncthreads();
  }
  if (threadIdx.x == 0) { npos[b] = sp[0]; nneg[b] = sn[0]; acc[b] = 0.0f; }
}

// ---------------------------------------------------------------------------
// Final scalar reduction -> cl_loss
// ---------------------------------------------------------------------------
__global__ void final_kernel(const float* __restrict__ acc,
                             const int* __restrict__ npos,
                             const int* __restrict__ nneg,
                             float* __restrict__ out) {
  const int b = threadIdx.x;  // 32 threads
  const int np = npos[b], nn = nneg[b];
  const float bl = acc[b] / (float)(np > 1 ? np : 1);
  const bool valid = (np >= 2) && (nn >= 1);
  float v = valid ? bl : 0.0f;
  float c = valid ? 1.0f : 0.0f;
#pragma unroll
  for (int off = 16; off > 0; off >>= 1) {
    v += __shfl_down(v, off, 32);
    c += __shfl_down(c, off, 32);
  }
  if (b == 0) out[0] = v / fmaxf(c, 1.0f);
}

// ---------------------------------------------------------------------------
extern "C" void kernel_launch(void* const* d_in, const int* in_sizes, int n_in,
                              void* d_out, int out_size, void* d_ws, size_t ws_size,
                              hipStream_t stream) {
  const float* pc       = (const float*)d_in[0];   // [32,1024,2]
  const float* pe_W     = (const float*)d_in[1];
  const float* pe_b     = (const float*)d_in[2];
  const float* type_emb = (const float*)d_in[3];
  const float* e_W1     = (const float*)d_in[4];
  const float* e_b1     = (const float*)d_in[5];
  const float* e_ln_w   = (const float*)d_in[6];
  const float* e_ln_b   = (const float*)d_in[7];
  const float* e_W2     = (const float*)d_in[8];
  const float* e_b2     = (const float*)d_in[9];
  const float* pj_W1    = (const float*)d_in[10];
  const float* pj_b1    = (const float*)d_in[11];
  const float* pj_ln_w  = (const float*)d_in[12];
  const float* pj_ln_b  = (const float*)d_in[13];
  const float* pj_W2    = (const float*)d_in[14];
  const float* pj_b2    = (const float*)d_in[15];
  const float* logT     = (const float*)d_in[16];
  const int*   labels   = (const int*)d_in[17];

  float* out_f32  = (float*)d_out;                   // prompt_embed [32,1024,256]
  float* out_loss = out_f32 + (size_t)M_TOTAL * EMBED_DIM;

  // workspace partitioning (256B aligned)
  size_t off = 0;
  auto salloc = [&](size_t bytes) -> void* {
    void* p = (char*)d_ws + off;
    off = (off + bytes + 255) & ~(size_t)255;
    return p;
  };
  __bf16* wPE   = (__bf16*)salloc((size_t)256 * 256 * 2);
  __bf16* wE1   = (__bf16*)salloc((size_t)256 * 256 * 2);
  __bf16* wE2   = (__bf16*)salloc((size_t)256 * 256 * 2);
  __bf16* wP1   = (__bf16*)salloc((size_t)256 * 256 * 2);
  __bf16* wP2   = (__bf16*)salloc((size_t)128 * 256 * 2);
  __bf16* bufA  = (__bf16*)salloc((size_t)M_TOTAL * 256 * 2);
  __bf16* bufB  = (__bf16*)salloc((size_t)M_TOTAL * 256 * 2);
  __bf16* projB = (__bf16*)salloc((size_t)M_TOTAL * 128 * 2);
  float*  acc   = (float*)salloc(NBATCH * sizeof(float));
  int*    npos  = (int*)salloc(NBATCH * sizeof(int));
  int*    nneg  = (int*)salloc(NBATCH * sizeof(int));

  // 1) weight converts
  f2bf_kernel<<<256, 256, 0, stream>>>(pe_W,  wPE, 256 * 256);
  f2bf_kernel<<<256, 256, 0, stream>>>(e_W1,  wE1, 256 * 256);
  f2bf_kernel<<<256, 256, 0, stream>>>(e_W2,  wE2, 256 * 256);
  f2bf_kernel<<<256, 256, 0, stream>>>(pj_W1, wP1, 256 * 256);
  f2bf_kernel<<<128, 256, 0, stream>>>(pj_W2, wP2, 128 * 256);

  // 2) positional-encoding features -> bufA
  pe_kernel<<<M_TOTAL * 256 / 256, 256, 0, stream>>>(pc, bufA);

  const int gemmGrid = M_TOTAL / 32;  // 1024 blocks

  // 3) pos_embed + type_emb[label]  -> bufB (bf16)
  gemm_ep_kernel<256, 0><<<gemmGrid, 256, 0, stream>>>(
      bufA, wPE, pe_b, nullptr, nullptr, type_emb, labels, nullptr, bufB);
  // 4) e_W1 + LN + GELU             -> bufA
  gemm_ep_kernel<256, 1><<<gemmGrid, 256, 0, stream>>>(
      bufB, wE1, e_b1, e_ln_w, e_ln_b, nullptr, nullptr, nullptr, bufA);
  // 5) e_W2 -> prompt_embed (f32 to d_out) + bf16 copy -> bufB
  gemm_ep_kernel<256, 2><<<gemmGrid, 256, 0, stream>>>(
      bufA, wE2, e_b2, nullptr, nullptr, nullptr, nullptr, out_f32, bufB);
  // 6) pj_W1 + LN + GELU            -> bufA
  gemm_ep_kernel<256, 1><<<gemmGrid, 256, 0, stream>>>(
      bufB, wP1, pj_b1, pj_ln_w, pj_ln_b, nullptr, nullptr, nullptr, bufA);
  // 7) pj_W2 + row L2 normalize     -> projB (bf16, [32768 x 128])
  gemm_ep_kernel<128, 3><<<gemmGrid, 256, 0, stream>>>(
      bufA, wP2, pj_b2, nullptr, nullptr, nullptr, nullptr, nullptr, projB);

  // 8) per-batch label counts + zero accumulators
  counts_kernel<<<NBATCH, 256, 0, stream>>>(labels, npos, nneg, acc);

  // 9) similarity + streaming masked LSE loss
  dim3 simGrid(NPTS / 32, NBATCH);
  sim_loss_kernel<<<simGrid, 256, 0, stream>>>(projB, labels, logT, acc);

  // 10) final scalar
  final_kernel<<<1, 32, 0, stream>>>(acc, npos, nneg, out_loss);
}
